// SemiEquivariantSparseStructureAttention_3685081940021
// MI455X (gfx1250) — matrix operand order repaired and checked by reading, not compiled, not run on hardware
//
#include <hip/hip_runtime.h>
#include <hip/hip_bf16.h>
#include <math.h>

// ---------------------------------------------------------------------------
// Problem constants (from reference)
// ---------------------------------------------------------------------------
#define NN    4096   // nodes
#define KK    64     // neighbours
#define CC    256    // local channels
#define CP    128    // pair channels
#define HEADS 8
#define SIZE  32
#define CATD  1792   // 8*128 + 8*32 + 8*48 + 8*16
#define QKVD  768    // 8*3*32

typedef __attribute__((ext_vector_type(16))) __bf16 v16bf;
typedef __attribute__((ext_vector_type(8)))  float  v8f;
typedef __attribute__((ext_vector_type(4)))  int    v4i;

// Optional CDNA5 async global->LDS path (guarded so compile never breaks)
#if defined(__AMDGCN__) && defined(__has_builtin)
#if __has_builtin(__builtin_amdgcn_global_load_async_to_lds_b128)
#define HAVE_ASYNC_LDS 1
#endif
#if __has_builtin(__builtin_amdgcn_s_wait_asynccnt)
#define HAVE_WAIT_ASYNC 1
#endif
#endif

#ifdef HAVE_ASYNC_LDS
// Builtin expects typed int4 pointers: AS1 (global) src, AS3 (LDS) dst.
typedef const __attribute__((address_space(1))) void* as1_cvp;
typedef __attribute__((address_space(3))) void*       as3_vp;
typedef __attribute__((address_space(1))) v4i*        as1_v4ip;
typedef __attribute__((address_space(3))) v4i*        as3_v4ip;
#endif

// ---------------------------------------------------------------------------
// bf16 WMMA GEMM:  C[M,N] = A[M,K] * B[K,N] (+ bias[N])
// Block: 256 threads = 8 waves; block tile 128(M) x 64(N); k-step 32.
// Wave (wm 0..3, wn 0..1) owns a 32x32 sub-tile: 2 A-frags x 2 B-frags ->
// 4 WMMAs per k-step (4x math per barrier, 2x fragment reuse).
// Staging is explicitly split into a batched-load phase and a cvt/store
// phase so all global_load_b128 issue before any wait (latency hiding).
// ---------------------------------------------------------------------------
#define LDA 40   // padded bf16 stride (80B = multiple of 16B)
#define LDB 40

__device__ __forceinline__ v16bf load_a_frag(const __bf16* As, int row, int half) {
    union { v16bf v; unsigned u[8]; } f;
    const unsigned* p = (const unsigned*)(As + row * LDA);
#pragma unroll
    for (int i = 0; i < 4; ++i) f.u[i]     = p[half * 4 + i];       // K = half*8 + 2i
#pragma unroll
    for (int i = 0; i < 4; ++i) f.u[4 + i] = p[8 + half * 4 + i];   // K = 16 + half*8 + 2i
    return f.v;
}

__device__ __forceinline__ v16bf load_b_frag(const __bf16* Bs, int col, int half) {
    union { v16bf v; unsigned u[8]; } f;
    const unsigned* p = (const unsigned*)(Bs + col * LDB);
#pragma unroll
    for (int i = 0; i < 4; ++i) f.u[i]     = p[half * 4 + i];       // K = half*8 + 2i
#pragma unroll
    for (int i = 0; i < 4; ++i) f.u[4 + i] = p[8 + half * 4 + i];   // K = 16 + half*8 + 2i
    return f.v;
}

__global__ __launch_bounds__(256)
void gemm_bf16(const float* __restrict__ A, const float* __restrict__ B,
               const float* __restrict__ bias, float* __restrict__ C,
               int M, int N, int K)
{
    __shared__ __bf16 As[128 * LDA];
    __shared__ __bf16 Bs[64 * LDB];

    const int t    = threadIdx.x;
    const int lane = t & 31;
    const int wave = t >> 5;
    const int wm   = wave >> 1;   // 0..3  (32-row strip)
    const int wn   = wave & 1;    // 0..1  (32-col strip)
    const int rowBase = blockIdx.y * 128;
    const int colBase = blockIdx.x * 64;

    const int half = lane >> 4;               // 0/1
    const int r0   = wm * 32 + (lane & 15);   // A frag rows
    const int c0   = wn * 32 + (lane & 15);   // B frag cols

    // Per-thread staging coordinates (constant across k-steps)
    const int a_mr[4] = { (t + 0*256) >> 3, (t + 1*256) >> 3,
                          (t + 2*256) >> 3, (t + 3*256) >> 3 };
    const int a_kc    = (t & 7) * 4;          // 8 float4 per 32-wide A row
    const int b_kr[2] = { (t + 0*256) >> 4, (t + 1*256) >> 4 };
    const int b_nc    = (t & 15) * 4;         // 16 float4 per 64-wide B row

    v8f acc00 = {}, acc01 = {}, acc10 = {}, acc11 = {};

    for (int kk = 0; kk < K; kk += 32) {
        // ---- batched load phase: 6 independent global_load_b128 ----
        float4 fa[4], fb[2];
#pragma unroll
        for (int q = 0; q < 4; ++q)
            fa[q] = *(const float4*)&A[(size_t)(rowBase + a_mr[q]) * K + kk + a_kc];
#pragma unroll
        for (int q = 0; q < 2; ++q)
            fb[q] = *(const float4*)&B[(size_t)(kk + b_kr[q]) * N + colBase + b_nc];

        // Hint-prefetch next k-tiles into cache (global_prefetch_b8)
        if (kk + 32 < K) {
            __builtin_prefetch(&A[(size_t)(rowBase + a_mr[0]) * K + kk + 32 + a_kc], 0, 0);
            __builtin_prefetch(&B[(size_t)(kk + 32 + b_kr[0]) * N + colBase + b_nc], 0, 0);
        }

        // ---- convert + store phase ----
#pragma unroll
        for (int q = 0; q < 4; ++q) {
            __bf16* d = &As[a_mr[q] * LDA + a_kc];
            d[0] = (__bf16)fa[q].x; d[1] = (__bf16)fa[q].y;
            d[2] = (__bf16)fa[q].z; d[3] = (__bf16)fa[q].w;
        }
#pragma unroll
        for (int q = 0; q < 2; ++q) {
            Bs[(b_nc + 0) * LDB + b_kr[q]] = (__bf16)fb[q].x;
            Bs[(b_nc + 1) * LDB + b_kr[q]] = (__bf16)fb[q].y;
            Bs[(b_nc + 2) * LDB + b_kr[q]] = (__bf16)fb[q].z;
            Bs[(b_nc + 3) * LDB + b_kr[q]] = (__bf16)fb[q].w;
        }
        __syncthreads();

        const v16bf a0 = load_a_frag(As, r0,      half);
        const v16bf a1 = load_a_frag(As, r0 + 16, half);
        const v16bf b0 = load_b_frag(Bs, c0,      half);
        const v16bf b1 = load_b_frag(Bs, c0 + 16, half);

        acc00 = __builtin_amdgcn_wmma_f32_16x16x32_bf16(false, a0, false, b0, (short)0, acc00, false, false);
        acc01 = __builtin_amdgcn_wmma_f32_16x16x32_bf16(false, a0, false, b1, (short)0, acc01, false, false);
        acc10 = __builtin_amdgcn_wmma_f32_16x16x32_bf16(false, a1, false, b0, (short)0, acc10, false, false);
        acc11 = __builtin_amdgcn_wmma_f32_16x16x32_bf16(false, a1, false, b1, (short)0, acc11, false, false);
        __syncthreads();
    }

    // Store D (f32 16x16 C/D layout): VGPR i -> M = i + half*8, N = lane&15
    const int n0 = colBase + wn * 32 + (lane & 15);
    const float bv0 = bias ? bias[n0]      : 0.f;
    const float bv1 = bias ? bias[n0 + 16] : 0.f;
#pragma unroll
    for (int i = 0; i < 8; ++i) {
        const int m0 = rowBase + wm * 32 + half * 8 + i;
        float* row0 = C + (size_t)m0 * N;
        float* row1 = C + (size_t)(m0 + 16) * N;
        row0[n0]      = acc00[i] + bv0;
        row0[n0 + 16] = acc01[i] + bv1;
        row1[n0]      = acc10[i] + bv0;
        row1[n0 + 16] = acc11[i] + bv1;
    }
}

// ---------------------------------------------------------------------------
// Prep: layernorm q/k, copy v, build geometric point tables.
// One block per node; wave = head; lane = channel.
// ---------------------------------------------------------------------------
__global__ __launch_bounds__(256)
void prep_kernel(const float* __restrict__ qkv, const float* __restrict__ pos,
                 const float* __restrict__ qsc, const float* __restrict__ qof,
                 const float* __restrict__ ksc, const float* __restrict__ kof,
                 float* __restrict__ qln, float* __restrict__ kln,
                 float* __restrict__ vv,  float* __restrict__ qg,
                 float* __restrict__ kg,  float* __restrict__ vg)
{
    const int i    = blockIdx.x;
    const int lane = threadIdx.x & 31;
    const int h    = threadIdx.x >> 5;
    const float* row = qkv + (size_t)i * QKVD + h * 96;
    const float pcarr[3] = {pos[i * 9 + 3], pos[i * 9 + 4], pos[i * 9 + 5]};

    // layernorm q
    {
        float x = row[lane];
        float s = x;
        for (int o = 16; o > 0; o >>= 1) s += __shfl_xor(s, o, 32);
        float mu = s * (1.f / 32.f);
        float d = x - mu, v2 = d * d;
        for (int o = 16; o > 0; o >>= 1) v2 += __shfl_xor(v2, o, 32);
        qln[((size_t)i * HEADS + h) * SIZE + lane] =
            d * rsqrtf(v2 * (1.f / 32.f) + 1e-5f) * qsc[lane] + qof[lane];
    }
    // layernorm k
    {
        float x = row[32 + lane];
        float s = x;
        for (int o = 16; o > 0; o >>= 1) s += __shfl_xor(s, o, 32);
        float mu = s * (1.f / 32.f);
        float d = x - mu, v2 = d * d;
        for (int o = 16; o > 0; o >>= 1) v2 += __shfl_xor(v2, o, 32);
        kln[((size_t)i * HEADS + h) * SIZE + lane] =
            d * rsqrtf(v2 * (1.f / 32.f) + 1e-5f) * ksc[lane] + kof[lane];
    }
    vv[((size_t)i * HEADS + h) * SIZE + lane] = row[64 + lane];

    if (lane < 24) {   // q_g: s=0..7, k_g: s=8..15 (24 comps each)
        qg[((size_t)i * HEADS + h) * 24 + lane] = row[lane]      + pcarr[lane % 3];
        kg[((size_t)i * HEADS + h) * 24 + lane] = row[24 + lane] + pcarr[lane % 3];
    }
    vg[((size_t)i * HEADS + h) * 48 + lane] = row[48 + lane] + pcarr[lane % 3];
    if (lane < 16) {
        int e = 32 + lane;
        vg[((size_t)i * HEADS + h) * 48 + e] = row[48 + e] + pcarr[e % 3];
    }
}

// ---------------------------------------------------------------------------
// Fused attention: one block per node, one wave per head.
// pair slab (64x128, 32KB) staged once in LDS (async global->LDS) and
// reused for bias logits AND out_pair (single HBM pass over `pair`).
// ---------------------------------------------------------------------------
__global__ __launch_bounds__(256)
void attn_kernel(const float* __restrict__ pair, const int* __restrict__ nbr,
                 const unsigned char* __restrict__ mask,
                 const float* __restrict__ Wb, const float* __restrict__ gamma,
                 const float* __restrict__ pos,
                 const float* __restrict__ qln, const float* __restrict__ kln,
                 const float* __restrict__ vv,  const float* __restrict__ qg,
                 const float* __restrict__ kg,  const float* __restrict__ vg,
                 float* __restrict__ feats)
{
    const int i    = blockIdx.x;
    const int t    = threadIdx.x;
    const int lane = t & 31;
    const int h    = t >> 5;

    __shared__ float pairS[KK * CP];       // 32 KB
    __shared__ float wbS[CP * HEADS];      // 4 KB
    __shared__ float qS[HEADS * SIZE];
    __shared__ float qgS[HEADS * 24];
    __shared__ int   nbrS[KK];
    __shared__ float mkS[KK];
    __shared__ float attnS[HEADS * KK];
    __shared__ float opS[HEADS * 48];

    const float* prow = pair + (size_t)i * KK * CP;
#ifdef HAVE_ASYNC_LDS
    {
        // async global->LDS, 16B per lane per issue (ASYNCcnt-tracked)
#pragma unroll
        for (int e = 0; e < KK * CP; e += 256 * 4) {
            as1_v4ip g = (as1_v4ip)(as1_cvp)(prow + t * 4 + e);
            as3_v4ip l = (as3_v4ip)(as3_vp)(&pairS[t * 4 + e]);
            __builtin_amdgcn_global_load_async_to_lds_b128(g, l, 0, 0);
        }
    }
#else
    for (int e = t; e < KK * CP; e += 256) pairS[e] = prow[e];
#endif
    for (int e = t; e < CP * HEADS; e += 256) wbS[e] = Wb[e];
    qS[t] = qln[(size_t)i * (HEADS * SIZE) + t];
    if (t < HEADS * 24) qgS[t] = qg[(size_t)i * (HEADS * 24) + t];
    if (t < KK) {
        int n = nbr[(size_t)i * KK + t];
        nbrS[t] = n;
        mkS[t] = (mask[(size_t)i * KK + t] != 0 && n != -1) ? 1.f : 0.f;
    }
#ifdef HAVE_ASYNC_LDS
#ifdef HAVE_WAIT_ASYNC
    __builtin_amdgcn_s_wait_asynccnt(0);
#else
    asm volatile("s_wait_asynccnt 0" ::: "memory");
#endif
#endif
    __syncthreads();

    const float wC2 = 0.5f * 0.16666667f;   // sqrt(2/(9*8)) / 2
    const float wL  = 0.57735026f;          // sqrt(1/3)
    const float sq  = 0.17677670f;          // sqrt(1/32)
    const float dfac = logf(1.f + expf(gamma[h])) * wC2;

    // ---- logits: lane handles j = lane and j = lane+32 ----
    float lg[2];
#pragma unroll
    for (int rep = 0; rep < 2; ++rep) {
        const int j = lane + rep * 32;
        const int n = nbrS[j];
        const float* pj = &pairS[j * CP];
        float b = 0.f;
#pragma unroll 8
        for (int c2 = 0; c2 < CP; ++c2) b = fmaf(pj[c2], wbS[c2 * HEADS + h], b);
        const float* kn = &kln[((size_t)n * HEADS + h) * SIZE];
        float dt = 0.f;
#pragma unroll 8
        for (int c2 = 0; c2 < SIZE; ++c2) dt = fmaf(qS[h * SIZE + c2], kn[c2], dt);
        const float* kgn = &kg[((size_t)n * HEADS + h) * 24];
        float ds = 0.f;
#pragma unroll 8
        for (int e = 0; e < 24; ++e) {
            float d = qgS[h * 24 + e] - kgn[e];
            ds = fmaf(d, d, ds);
        }
        float l = wL * (sq * dt + b - dfac * ds);
        lg[rep] = (mkS[j] > 0.f) ? l : -1e9f;
    }

    // ---- softmax over K=64 within the wave ----
    float m = fmaxf(lg[0], lg[1]);
    for (int o = 16; o > 0; o >>= 1) m = fmaxf(m, __shfl_xor(m, o, 32));
    float e0 = expf(lg[0] - m), e1 = expf(lg[1] - m);
    float s = e0 + e1;
    for (int o = 16; o > 0; o >>= 1) s += __shfl_xor(s, o, 32);
    const float inv = 1.f / s;
    attnS[h * KK + lane]      = e0 * inv * mkS[lane];
    attnS[h * KK + lane + 32] = e1 * inv * mkS[lane + 32];
    // attnS/opS are wave-local: LDS ops within a wave are in-order.

    float* fr = feats + (size_t)i * CATD;

    // ---- out_pair: 128 channels, 4 per lane, pair from LDS ----
#pragma unroll
    for (int rep = 0; rep < 4; ++rep) {
        const int c2 = lane + rep * 32;
        float acc = 0.f;
        for (int j = 0; j < KK; ++j)
            acc = fmaf(attnS[h * KK + j], pairS[j * CP + c2], acc);
        fr[h * CP + c2] = acc;
    }

    // ---- out_scalar: 32 channels (L2-resident gather of v rows) ----
    {
        float acc = 0.f;
        for (int j = 0; j < KK; ++j)
            acc = fmaf(attnS[h * KK + j],
                       vv[((size_t)nbrS[j] * HEADS + h) * SIZE + lane], acc);
        fr[1024 + h * SIZE + lane] = acc;
    }

    // ---- out_point: 48 channels, then norms ----
    const float pc[3] = {pos[i * 9 + 3], pos[i * 9 + 4], pos[i * 9 + 5]};
#pragma unroll
    for (int rep = 0; rep < 2; ++rep) {
        const int e = lane + rep * 32;
        if (e < 48) {
            float acc = 0.f;
            for (int j = 0; j < KK; ++j)
                acc = fmaf(attnS[h * KK + j],
                           vg[((size_t)nbrS[j] * HEADS + h) * 48 + e], acc);
            acc -= pc[e % 3];
            opS[h * 48 + e] = acc;
            fr[1280 + h * 48 + e] = acc;
        }
    }
    if (lane < 16) {
        float x = opS[h * 48 + lane * 3 + 0];
        float y = opS[h * 48 + lane * 3 + 1];
        float z = opS[h * 48 + lane * 3 + 2];
        fr[1664 + h * 16 + lane] = sqrtf(fmaxf(x * x + y * y + z * z, 1e-6f));
    }
}

// ---------------------------------------------------------------------------
// Launch
// ---------------------------------------------------------------------------
extern "C" void kernel_launch(void* const* d_in, const int* in_sizes, int n_in,
                              void* d_out, int out_size, void* d_ws, size_t ws_size,
                              hipStream_t stream)
{
    const float* local = (const float*)d_in[0];
    const float* pair  = (const float*)d_in[1];
    const float* pos   = (const float*)d_in[2];
    const int*   nbr   = (const int*)d_in[3];
    const unsigned char* mask = (const unsigned char*)d_in[4];
    const float* W_qkv = (const float*)d_in[5];
    const float* qsc   = (const float*)d_in[6];
    const float* qof   = (const float*)d_in[7];
    const float* ksc   = (const float*)d_in[8];
    const float* kof   = (const float*)d_in[9];
    const float* Wb    = (const float*)d_in[10];
    const float* gamma = (const float*)d_in[11];
    const float* W_out = (const float*)d_in[12];
    const float* b_out = (const float*)d_in[13];
    float* out = (float*)d_out;

    // Workspace carve-up (all f32)
    float* ws   = (float*)d_ws;
    float* qkv  = ws;  ws += (size_t)NN * QKVD;
    float* qln  = ws;  ws += (size_t)NN * HEADS * SIZE;
    float* kln  = ws;  ws += (size_t)NN * HEADS * SIZE;
    float* vv   = ws;  ws += (size_t)NN * HEADS * SIZE;
    float* qg   = ws;  ws += (size_t)NN * HEADS * 24;
    float* kg   = ws;  ws += (size_t)NN * HEADS * 24;
    float* vg   = ws;  ws += (size_t)NN * HEADS * 48;
    float* feats= ws;  ws += (size_t)NN * CATD;

    // 1) qkv = local @ W_qkv   (M=4096, N=768, K=256)
    gemm_bf16<<<dim3(QKVD / 64, NN / 128), 256, 0, stream>>>(
        local, W_qkv, nullptr, qkv, NN, QKVD, CC);

    // 2) layernorm + geometric point prep
    prep_kernel<<<NN, 256, 0, stream>>>(
        qkv, pos, qsc, qof, ksc, kof, qln, kln, vv, qg, kg, vg);

    // 3) fused attention -> feats (single HBM pass over `pair`)
    attn_kernel<<<NN, 256, 0, stream>>>(
        pair, nbr, mask, Wb, gamma, pos, qln, kln, vv, qg, kg, vg, feats);

    // 4) out = feats @ W_out + b_out   (M=4096, N=256, K=1792)
    gemm_bf16<<<dim3(CC / 64, NN / 128), 256, 0, stream>>>(
        feats, W_out, b_out, out, NN, CC, CATD);

    (void)in_sizes; (void)n_in; (void)out_size; (void)ws_size;
}
